// AttentionBasedMemory_32263794327941
// MI455X (gfx1250) — matrix operand order repaired and checked by reading, not compiled, not run on hardware
//
#include <hip/hip_runtime.h>
#include <math.h>

// ---------------------------------------------------------------------------
// Problem constants (match reference)
// ---------------------------------------------------------------------------
#define B_Q    2048
#define M_KV   65536
#define D_DIM  1024
#define TOPK   64
#define THRESH 2e-5f
#define SCALE  0.03125f   // 1/sqrt(1024)

// ---------------------------------------------------------------------------
// Tiling
// ---------------------------------------------------------------------------
#define ROWS  32           // query rows per block (2 x 16-row WMMA tiles)
#define MT    128          // keys per M-tile
#define DST   256          // D elements per LDS stage (4 stages / tile)
#define NSTG  (D_DIM / DST)        // 4
#define TPB   128          // 4 waves (wave32)
#define SPLIT 8            // M split for parallelism
#define TILES_PER_SPLIT ((M_KV / MT) / SPLIT)   // 64
#define NSCAN 4            // scanner threads per row (TPB / ROWS)

#define QSTR 1032          // bf16 elems per Q row in LDS (pad -> 16B aligned)
#define KSTR 264           // bf16 elems per K row in LDS (pad -> 16B aligned)
#define SSTR 132           // f32 elems per score row in LDS

typedef __bf16 bf16t;
typedef __attribute__((ext_vector_type(16))) bf16t v16bf;
typedef __attribute__((ext_vector_type(8)))  bf16t bf16x8;
typedef __attribute__((ext_vector_type(4)))  bf16t bf16x4;
typedef __attribute__((ext_vector_type(8)))  float v8f;
typedef __attribute__((ext_vector_type(4)))  float f4;

struct RowStats { float m, z; };

#define QS_BYTES ((size_t)ROWS * QSTR * 2)
#define KS_ONE   ((size_t)MT * KSTR * 2)          // one K buffer (67.6 KB)
#define KS_BYTES (2 * KS_ONE)                     // double buffered
#define SM_BASE  ((size_t)(QS_BYTES + KS_BYTES + ROWS*SSTR*4 + ROWS*NSCAN*4 + ROWS*NSCAN*4))
#define SM_P2    ((size_t)(SM_BASE + ROWS*NSCAN*TOPK*8))

// ---------------------------------------------------------------------------
// CDNA5 async global->LDS copy (ASYNCcnt path, no VGPR round trip)
// ---------------------------------------------------------------------------
__device__ __forceinline__ void async_copy_b128(unsigned lds_off, const void* gaddr) {
    asm volatile("global_load_async_to_lds_b128 %0, %1, off"
                 :: "v"(lds_off), "v"(gaddr) : "memory");
}
__device__ __forceinline__ void wait_async_32() {
    asm volatile("s_wait_asynccnt 32" ::: "memory");    // prev stage done, next in flight
}
__device__ __forceinline__ void wait_async_0() {
    asm volatile("s_wait_asynccnt 0" ::: "memory");
}
__device__ __forceinline__ void wait_ds_0() {
    asm volatile("s_wait_dscnt 0" ::: "memory");        // drain ds reads before async WAR
}

// ---------------------------------------------------------------------------
// One-time fp32 -> bf16 conversion of Q and K into workspace.
// After this, K_bf16 (128 MB) is L2-resident (192 MB L2) for both GEMM passes.
// ---------------------------------------------------------------------------
__global__ __launch_bounds__(256)
void convert_bf16(const float* __restrict__ Q, const float* __restrict__ Kmem,
                  bf16t* __restrict__ Qbf, bf16t* __restrict__ Kbf)
{
    const size_t nq = (size_t)B_Q * D_DIM / 4;
    const size_t nk = (size_t)M_KV * D_DIM / 4;
    size_t u = (size_t)blockIdx.x * 256 + threadIdx.x;
    const float* src; bf16t* dst; size_t i;
    if (u < nq)            { src = Q;    dst = Qbf; i = u; }
    else if (u < nq + nk)  { src = Kmem; dst = Kbf; i = u - nq; }
    else return;
    f4 v = ((const f4*)src)[i];
    bf16x4 w; w[0] = (bf16t)v.x; w[1] = (bf16t)v.y; w[2] = (bf16t)v.z; w[3] = (bf16t)v.w;
    *(bf16x4*)(dst + i * 4) = w;
}

// ---------------------------------------------------------------------------
// PASS 1: exact per-row (max, Z) over this block's M split.
// PASS 2: masked sum + per-row top-64 candidates over this block's M split.
// KBF = true: bf16 operands in ws; async double-buffered LDS staging.
// ---------------------------------------------------------------------------
template <int PASS, bool KBF>
__global__ __launch_bounds__(TPB)
void attn_pass(const float* __restrict__ Q, const float* __restrict__ Kmem,
               const bf16t* __restrict__ Qbf, const bf16t* __restrict__ Kbf,
               RowStats* __restrict__ statsP, const RowStats* __restrict__ stats,
               float* __restrict__ msumP,
               float* __restrict__ candS, int* __restrict__ candI)
{
    extern __shared__ char smem[];
    bf16t* Qs   = (bf16t*)smem;                           // [32][QSTR] bf16
    bf16t* Ks   = (bf16t*)(smem + QS_BYTES);              // 2 x [128][KSTR] bf16
    float* Ssc  = (float*)(smem + QS_BYTES + KS_BYTES);   // [32][SSTR] f32
    float* red  = Ssc + ROWS * SSTR;                      // [32][4]
    int*   cntA = (int*)(red + ROWS * NSCAN);             // [32][4]
    float* listS = (float*)(cntA + ROWS * NSCAN);         // PASS2: [32][4][64]
    int*   listI = (int*)(listS + ROWS * NSCAN * TOPK);

    const int t    = threadIdx.x;
    const int wave = t >> 5;
    const int lane = t & 31;
    const int rb = blockIdx.x / SPLIT;
    const int sp = blockIdx.x % SPLIT;
    const int qbase = rb * ROWS;
    const int r = t >> 2;          // stats: row 0..31
    const int j = t & 3;           // stats: 4 scanner threads per row
    const unsigned ldsbase = (unsigned)__builtin_amdgcn_groupstaticsize();
    const int mt0 = sp * TILES_PER_SPLIT;

    // issue one K stage (32 async b128 per thread) into buffer `buf`
    auto issue_stage = [&](int mt, int stg, int buf) {
        const char* kg = (const char*)(Kbf + (size_t)mt * MT * D_DIM + stg * DST);
        unsigned base = ldsbase + (unsigned)QS_BYTES + (unsigned)(buf * KS_ONE);
        #pragma unroll 4
        for (int it = 0; it < MT * (DST * 2 / 16) / TPB; ++it) {     // 32 iters
            int u = t + it * TPB;
            int kr = u >> 5, c16 = u & 31;       // 32 x 16B units per row
            async_copy_b128(base + (unsigned)(kr * KSTR * 2 + c16 * 16),
                            kg + (size_t)kr * (D_DIM * 2) + c16 * 16);
        }
    };

    // --- stage Q block into LDS + prologue K stage ----------------------
    if (KBF) {
        #pragma unroll 4
        for (int it = 0; it < ROWS * 128 / TPB; ++it) {      // 32 iters, 16B units
            int u = t + it * TPB;
            int qr = u >> 7, c16 = u & 127;
            async_copy_b128(ldsbase + (unsigned)(qr * QSTR * 2 + c16 * 16),
                            (const char*)(Qbf + (size_t)(qbase + qr) * D_DIM) + c16 * 16);
        }
        issue_stage(mt0, 0, 0);     // stage g=0 -> buf0 (in order after Q copy)
    } else {
        for (int it = 0; it < ROWS * (D_DIM / 4) / TPB; it += 8) {  // 64 iters
            f4 v[8];
            #pragma unroll
            for (int b2 = 0; b2 < 8; ++b2) {
                int u = t + (it + b2) * TPB;
                int qr = u >> 8, c4 = u & 255;
                v[b2] = ((const f4*)(Q + (size_t)(qbase + qr) * D_DIM))[c4];
            }
            #pragma unroll
            for (int b2 = 0; b2 < 8; ++b2) {
                int u = t + (it + b2) * TPB;
                int qr = u >> 8, c4 = u & 255;
                bf16x4 w; w[0]=(bf16t)v[b2].x; w[1]=(bf16t)v[b2].y;
                          w[2]=(bf16t)v[b2].z; w[3]=(bf16t)v[b2].w;
                *(bf16x4*)(Qs + qr * QSTR + c4 * 4) = w;
            }
        }
    }

    float run_m = -3.0e38f, run_z = 0.0f;                 // PASS 1
    float row_m = 0.f, row_z = 1.f, sthr = 3.0e38f;       // PASS 2
    float msum = 0.f;
    int   cnt = 0; float cmin = -3.0e38f;
    float* myS = listS + (r * NSCAN + j) * TOPK;
    int*   myI = listI + (r * NSCAN + j) * TOPK;

    if (PASS == 2) {
        RowStats st = stats[qbase + r];
        row_m = st.m; row_z = st.z;
        sthr = row_m + __logf(THRESH) + __logf(row_z);    // attn>t  <=>  s>sthr
    }

    for (int mti = 0; mti < TILES_PER_SPLIT; ++mti) {
        const int mt = mt0 + mti;
        v8f acc00, acc01, acc10, acc11;
        #pragma unroll
        for (int ii = 0; ii < 8; ++ii) {
            acc00[ii] = 0.f; acc01[ii] = 0.f; acc10[ii] = 0.f; acc11[ii] = 0.f;
        }

        for (int stg = 0; stg < NSTG; ++stg) {
            const int g = mti * NSTG + stg;
            const int buf = KBF ? (g & 1) : 0;
            __syncthreads();   // all waves done computing on the buffer we fill next
            if (KBF) {
                wait_ds_0();   // our ds reads of that buffer are drained (WAR vs async)
                const int gn = g + 1;
                if (gn < TILES_PER_SPLIT * NSTG) {
                    issue_stage(mt0 + gn / NSTG, gn % NSTG, gn & 1);
                    wait_async_32();   // in-order: <=32 outstanding => stage g landed
                } else {
                    wait_async_0();
                }
            } else {
                const float* kg = Kmem + (size_t)mt * MT * D_DIM + stg * DST;
                if (stg == 0 && mti + 1 < TILES_PER_SPLIT)
                    __builtin_prefetch(Kmem + (size_t)(mt + 1) * MT * D_DIM + t * 256, 0, 1);
                for (int it = 0; it < MT * (DST / 4) / TPB; it += 8) {  // 64 iters
                    f4 v[8];
                    #pragma unroll
                    for (int b2 = 0; b2 < 8; ++b2) {
                        int u = t + (it + b2) * TPB;
                        int kr = u >> 6, c4 = u & 63;
                        v[b2] = ((const f4*)(kg + (size_t)kr * D_DIM))[c4];
                    }
                    #pragma unroll
                    for (int b2 = 0; b2 < 8; ++b2) {
                        int u = t + (it + b2) * TPB;
                        int kr = u >> 6, c4 = u & 63;
                        bf16x4 w; w[0]=(bf16t)v[b2].x; w[1]=(bf16t)v[b2].y;
                                  w[2]=(bf16t)v[b2].z; w[3]=(bf16t)v[b2].w;
                        *(bf16x4*)(Ks + kr * KSTR + c4 * 4) = w;
                    }
                }
            }
            __syncthreads();   // stage g visible to all waves

            // --- fragment base pointers hoisted: per-chunk addresses are
            // --- base + <16-bit immediate>, so no VALU writes interleave
            // --- with WMMA issue (avoids WMMA->VALU WAR hazard NOPs).
            const bf16t* Kb = Ks + (size_t)buf * (MT * KSTR);
            const int keyb = wave * 32;
            const int half = lane >> 4, mrow = lane & 15;
            const int n = lane & 15,   kh = lane >> 4;
            const bf16t* qa0 = Qs + mrow * QSTR + stg * DST + half * 8;   // rows 0-15
            const bf16t* qa1 = qa0 + 16 * QSTR;                           // rows 16-31
            const bf16t* kb0 = Kb + (keyb + n) * KSTR + kh * 8;           // keys +0..15
            const bf16t* kb1 = kb0 + 16 * KSTR;                           // keys +16..31
            #pragma unroll
            for (int cc = 0; cc < DST / 32; ++cc) {               // 8 chunks
                const int o = cc * 32;                            // elems (64B steps)
                bf16x8 a0lo = *(const bf16x8*)(qa0 + o);
                bf16x8 a0hi = *(const bf16x8*)(qa0 + o + 16);
                bf16x8 a1lo = *(const bf16x8*)(qa1 + o);
                bf16x8 a1hi = *(const bf16x8*)(qa1 + o + 16);
                v16bf a0, a1;
                #pragma unroll
                for (int ii = 0; ii < 8; ++ii) {
                    a0[ii] = a0lo[ii]; a0[ii + 8] = a0hi[ii];
                    a1[ii] = a1lo[ii]; a1[ii + 8] = a1hi[ii];
                }
                bf16x8 b0lo = *(const bf16x8*)(kb0 + o);
                bf16x8 b0hi = *(const bf16x8*)(kb0 + o + 16);
                bf16x8 b1lo = *(const bf16x8*)(kb1 + o);
                bf16x8 b1hi = *(const bf16x8*)(kb1 + o + 16);
                v16bf b0, b1;
                #pragma unroll
                for (int ii = 0; ii < 8; ++ii) {
                    b0[ii] = b0lo[ii]; b0[ii + 8] = b0hi[ii];
                    b1[ii] = b1lo[ii]; b1[ii + 8] = b1hi[ii];
                }
                acc00 = __builtin_amdgcn_wmma_f32_16x16x32_bf16(false, a0, false, b0, (short)0, acc00, false, false);
                acc01 = __builtin_amdgcn_wmma_f32_16x16x32_bf16(false, a0, false, b1, (short)0, acc01, false, false);
                acc10 = __builtin_amdgcn_wmma_f32_16x16x32_bf16(false, a1, false, b0, (short)0, acc10, false, false);
                acc11 = __builtin_amdgcn_wmma_f32_16x16x32_bf16(false, a1, false, b1, (short)0, acc11, false, false);
            }
        }

        __syncthreads();   // previous tile's Ssc readers done
        {   // C/D layout: lane<16 -> M=v,N=lane ; lane>=16 -> M=8+v,N=lane-16
            const int n2 = lane & 15, rbank = (lane >> 4) * 8, keyb = wave * 32;
            #pragma unroll
            for (int v = 0; v < 8; ++v) {
                Ssc[(rbank + v) * SSTR + keyb + n2]           = acc00[v] * SCALE;
                Ssc[(rbank + v) * SSTR + keyb + 16 + n2]      = acc01[v] * SCALE;
                Ssc[(16 + rbank + v) * SSTR + keyb + n2]      = acc10[v] * SCALE;
                Ssc[(16 + rbank + v) * SSTR + keyb + 16 + n2] = acc11[v] * SCALE;
            }
        }
        __syncthreads();

        const float* sr = Ssc + r * SSTR + j * 32;
        if (PASS == 1) {
            float pmax = -3.0e38f;
            #pragma unroll
            for (int ii = 0; ii < 32; ++ii) pmax = fmaxf(pmax, sr[ii]);
            red[r * NSCAN + j] = pmax;
            __syncthreads();
            float tmax = red[r * NSCAN];
            #pragma unroll
            for (int ii = 1; ii < NSCAN; ++ii) tmax = fmaxf(tmax, red[r * NSCAN + ii]);
            float ps = 0.f;
            #pragma unroll
            for (int ii = 0; ii < 32; ++ii) ps += __expf(sr[ii] - tmax);
            __syncthreads();
            red[r * NSCAN + j] = ps;
            __syncthreads();
            if (j == 0) {
                float ts = 0.f;
                #pragma unroll
                for (int ii = 0; ii < NSCAN; ++ii) ts += red[r * NSCAN + ii];
                if (tmax > run_m) { run_z = run_z * __expf(run_m - tmax) + ts; run_m = tmax; }
                else               run_z += ts * __expf(tmax - run_m);
            }
        } else {
            const int colbase = mt * MT + j * 32;
            for (int ii = 0; ii < 32; ++ii) {
                float s = sr[ii];
                if (s > sthr) msum += __expf(s - row_m);
                if (cnt < TOPK) {                         // sorted-descending insert
                    int p = cnt++;
                    while (p > 0 && myS[p - 1] < s) { myS[p] = myS[p - 1]; myI[p] = myI[p - 1]; --p; }
                    myS[p] = s; myI[p] = colbase + ii;
                    cmin = myS[cnt - 1];
                } else if (s > cmin) {
                    int p = TOPK - 1;
                    while (p > 0 && myS[p - 1] < s) { myS[p] = myS[p - 1]; myI[p] = myI[p - 1]; --p; }
                    myS[p] = s; myI[p] = colbase + ii;
                    cmin = myS[TOPK - 1];
                }
            }
        }
    }

    if (PASS == 1) {
        if (j == 0) {
            RowStats st; st.m = run_m; st.z = run_z;
            statsP[(qbase + r) * SPLIT + sp] = st;
        }
    } else {
        red[r * NSCAN + j]  = msum;
        cntA[r * NSCAN + j] = cnt;
        __syncthreads();
        if (j == 0) {
            float mtot = 0.f;
            #pragma unroll
            for (int ii = 0; ii < NSCAN; ++ii) mtot += red[r * NSCAN + ii];
            msumP[(qbase + r) * SPLIT + sp] = mtot;
            int head[NSCAN];
            #pragma unroll
            for (int w = 0; w < NSCAN; ++w) head[w] = 0;
            float* oS = candS + (size_t)((qbase + r) * SPLIT + sp) * TOPK;
            int*   oI = candI + (size_t)((qbase + r) * SPLIT + sp) * TOPK;
            for (int o = 0; o < TOPK; ++o) {
                float best = -3.0e38f; int bw = 0;
                #pragma unroll
                for (int w = 0; w < NSCAN; ++w) {
                    if (head[w] < cntA[r * NSCAN + w]) {
                        float v = listS[(r * NSCAN + w) * TOPK + head[w]];
                        if (v > best) { best = v; bw = w; }
                    }
                }
                int gi = listI[(r * NSCAN + bw) * TOPK + head[bw]];
                head[bw]++;
                oS[o] = best; oI[o] = gi;
            }
        }
    }
}

// ---------------------------------------------------------------------------
__global__ void combine_stats(const RowStats* __restrict__ statsP,
                              RowStats* __restrict__ stats)
{
    int row = blockIdx.x * blockDim.x + threadIdx.x;
    if (row >= B_Q) return;
    float m = -3.0e38f;
    RowStats p[SPLIT];
    #pragma unroll
    for (int s = 0; s < SPLIT; ++s) { p[s] = statsP[row * SPLIT + s]; m = fmaxf(m, p[s].m); }
    float z = 0.f;
    #pragma unroll
    for (int s = 0; s < SPLIT; ++s) z += p[s].z * __expf(p[s].m - m);
    RowStats o; o.m = m; o.z = z;
    stats[row] = o;
}

// ---------------------------------------------------------------------------
__global__ void finalize_topk(const RowStats* __restrict__ stats,
                              const float* __restrict__ msumP,
                              const float* __restrict__ candS,
                              const int* __restrict__ candI,
                              int* __restrict__ idxOut, float* __restrict__ wOut)
{
    int row = blockIdx.x * blockDim.x + threadIdx.x;
    if (row >= B_Q) return;
    RowStats st = stats[row];
    float mtot = 0.f;
    #pragma unroll
    for (int s = 0; s < SPLIT; ++s) mtot += msumP[row * SPLIT + s];
    float den  = mtot / st.z + 1e-8f;
    float sthr = st.m + __logf(THRESH) + __logf(st.z);
    int head[SPLIT];
    #pragma unroll
    for (int s = 0; s < SPLIT; ++s) head[s] = 0;
    const float* cs = candS + (size_t)row * SPLIT * TOPK;
    const int*   ci = candI + (size_t)row * SPLIT * TOPK;
    for (int o = 0; o < TOPK; ++o) {
        float best = -3.0e38f; int bw = 0;
        #pragma unroll
        for (int s = 0; s < SPLIT; ++s) {
            if (head[s] < TOPK) {
                float v = cs[s * TOPK + head[s]];
                if (v > best) { best = v; bw = s; }
            }
        }
        int gi = ci[bw * TOPK + head[bw]];
        head[bw]++;
        float p   = __expf(best - st.m) / st.z;
        float wgt = (best > sthr) ? (p / den) : 0.f;
        idxOut[row * TOPK + o] = gi;
        wOut[row * TOPK + o]   = wgt;
    }
}

// ---------------------------------------------------------------------------
// out layout: keys[B][64][D] | values[B][64][D] | weights[B][64]
// ---------------------------------------------------------------------------
__global__ __launch_bounds__(256)
void gather_out(const float* __restrict__ Kmem, const float* __restrict__ Vmem,
                const int* __restrict__ idx, const float* __restrict__ w,
                float* __restrict__ out)
{
    const size_t valsOff = (size_t)B_Q * TOPK * D_DIM;
    const int wave = threadIdx.x >> 5, lane = threadIdx.x & 31;
    const int slot = blockIdx.x * 8 + wave;        // b*64 + k
    const int id = idx[slot];
    const f4* ks = (const f4*)(Kmem + (size_t)id * D_DIM);
    const f4* vs = (const f4*)(Vmem + (size_t)id * D_DIM);
    f4* ko = (f4*)(out + (size_t)slot * D_DIM);
    f4* vo = (f4*)(out + valsOff + (size_t)slot * D_DIM);
    #pragma unroll 4
    for (int i = lane; i < D_DIM / 4; i += 32) { ko[i] = ks[i]; vo[i] = vs[i]; }
    if (lane == 0) out[2 * valsOff + slot] = w[slot];
}

// ---------------------------------------------------------------------------
extern "C" void kernel_launch(void* const* d_in, const int* in_sizes, int n_in,
                              void* d_out, int out_size, void* d_ws, size_t ws_size,
                              hipStream_t stream)
{
    (void)in_sizes; (void)n_in; (void)out_size;
    const float* Q    = (const float*)d_in[0];
    const float* Kmem = (const float*)d_in[1];
    const float* Vmem = (const float*)d_in[2];
    float* out = (float*)d_out;

    char* ws = (char*)d_ws;
    size_t off = 0;
    auto carve = [&](size_t bytes) -> char* {
        char* p = ws + off;
        off = (off + bytes + 255) & ~(size_t)255;
        return p;
    };
    RowStats* statsP = (RowStats*)carve((size_t)B_Q * SPLIT * sizeof(RowStats));
    RowStats* stats  = (RowStats*)carve((size_t)B_Q * sizeof(RowStats));
    float* msumP = (float*)carve((size_t)B_Q * SPLIT * sizeof(float));
    float* candS = (float*)carve((size_t)B_Q * SPLIT * TOPK * sizeof(float));
    int*   candI = (int*)carve((size_t)B_Q * SPLIT * TOPK * sizeof(int));
    int*   idxO  = (int*)carve((size_t)B_Q * TOPK * sizeof(int));
    float* wO    = (float*)carve((size_t)B_Q * TOPK * sizeof(float));
    bf16t* Qbf   = (bf16t*)carve((size_t)B_Q * D_DIM * sizeof(bf16t));
    bf16t* Kbf   = (bf16t*)carve((size_t)M_KV * D_DIM * sizeof(bf16t));
    const bool useBf = (ws_size >= off);     // ~142 MB for the L2-resident bf16-K path

    hipFuncSetAttribute(reinterpret_cast<const void*>(attn_pass<1, true>),
                        hipFuncAttributeMaxDynamicSharedMemorySize, (int)SM_P2);
    hipFuncSetAttribute(reinterpret_cast<const void*>(attn_pass<2, true>),
                        hipFuncAttributeMaxDynamicSharedMemorySize, (int)SM_P2);
    hipFuncSetAttribute(reinterpret_cast<const void*>(attn_pass<1, false>),
                        hipFuncAttributeMaxDynamicSharedMemorySize, (int)SM_P2);
    hipFuncSetAttribute(reinterpret_cast<const void*>(attn_pass<2, false>),
                        hipFuncAttributeMaxDynamicSharedMemorySize, (int)SM_P2);

    dim3 g((B_Q / ROWS) * SPLIT), b(TPB);
    if (useBf) {
        size_t units = ((size_t)B_Q + M_KV) * D_DIM / 4;
        convert_bf16<<<(unsigned)((units + 255) / 256), 256, 0, stream>>>(Q, Kmem, Qbf, Kbf);
        attn_pass<1, true><<<g, b, SM_BASE, stream>>>(Q, Kmem, Qbf, Kbf, statsP, stats, msumP, candS, candI);
        combine_stats<<<(B_Q + 255) / 256, 256, 0, stream>>>(statsP, stats);
        attn_pass<2, true><<<g, b, SM_P2, stream>>>(Q, Kmem, Qbf, Kbf, statsP, stats, msumP, candS, candI);
    } else {
        attn_pass<1, false><<<g, b, SM_BASE, stream>>>(Q, Kmem, Qbf, Kbf, statsP, stats, msumP, candS, candI);
        combine_stats<<<(B_Q + 255) / 256, 256, 0, stream>>>(statsP, stats);
        attn_pass<2, false><<<g, b, SM_P2, stream>>>(Q, Kmem, Qbf, Kbf, statsP, stats, msumP, candS, candI);
    }
    finalize_topk<<<(B_Q + 255) / 256, 256, 0, stream>>>(stats, msumP, candS, candI, idxO, wO);
    gather_out<<<B_Q * TOPK / 8, 256, 0, stream>>>(Kmem, Vmem, idxO, wO, out);
}
